// EncoderLayer_46969762349729
// MI455X (gfx1250) — matrix operand order repaired and checked by reading, not compiled, not run on hardware
//
#include <hip/hip_runtime.h>
#include <hip/hip_bf16.h>

typedef __attribute__((ext_vector_type(16))) _Float16 v16h;
typedef __attribute__((ext_vector_type(8)))  _Float16 v8h;
typedef __attribute__((ext_vector_type(8)))  float    v8f;

#define LSEQ 1024
#define CCH  1024
#define NB   4
#define NHH  32
#define EE   32
#define MROWS 4096          // NB*LSEQ
#define PSTR 264            // padded stride for Qp/Kp (257 valid cols)
#define NDIST 257

__device__ __forceinline__ v16h mk16(v8h lo, v8h hi) {
  v16h r;
#pragma unroll
  for (int i = 0; i < 8; ++i) { r[i] = lo[i]; r[i + 8] = hi[i]; }
  return r;
}

__device__ __forceinline__ v8f wmma_f16(v16h a, v16h b, v8f c) {
  return __builtin_amdgcn_wmma_f32_16x16x32_f16(false, a, false, b, (short)0, c,
                                                false, false);
}

__device__ __forceinline__ v8f zero8f() {
  v8f z = {0.f, 0.f, 0.f, 0.f, 0.f, 0.f, 0.f, 0.f};
  return z;
}

// ---------------- fp32 -> f16 convert (with zero padding tail) ----------------
__global__ void k_cvt(const float* __restrict__ s, _Float16* __restrict__ d,
                      int n, int ntot) {
  int i = blockIdx.x * blockDim.x + threadIdx.x;
  if (i < ntot) d[i] = (i < n) ? (_Float16)s[i] : (_Float16)0.f;
}

// ------- fp32 (CxC) -> f16 transposed, LDS-tiled (both sides coalesced) ------
__global__ __launch_bounds__(256) void k_trans(const float* __restrict__ W,
                                               _Float16* __restrict__ WT) {
  __shared__ float tile[32][33];              // +1 pad: no bank conflicts
  int bx = blockIdx.x & 31, by = blockIdx.x >> 5;
  int tx = threadIdx.x & 31, ty = threadIdx.x >> 5;   // 32 x 8
#pragma unroll
  for (int i = 0; i < 4; ++i)
    tile[ty + 8 * i][tx] = W[(size_t)(by * 32 + ty + 8 * i) * CCH + bx * 32 + tx];
  __syncthreads();
#pragma unroll
  for (int i = 0; i < 4; ++i)
    WT[(size_t)(bx * 32 + ty + 8 * i) * CCH + by * 32 + tx] =
        (_Float16)tile[tx][ty + 8 * i];
}

// ---------------- GEMM: (4096x1024) x (1024x1024), B given transposed --------
// one wave computes a 32x64 output tile (8 wmma accumulators; B frags reused
// across two A row-fragments -> 2x arithmetic intensity on the B stream)
// mode 0: f16 out, layout (N,H,L,E)      (Q, K)
// mode 1: f16 out, layout (N,H,E,L)      (V transposed)
// mode 2: relu + f16 out, row-major      (MLP mid)
// mode 3: f32 out, row-major             (MLP out)
__global__ __launch_bounds__(256) void k_gemm(const _Float16* __restrict__ A,
                                              const _Float16* __restrict__ BT,
                                              const float* __restrict__ bias,
                                              void* __restrict__ outp, int mode) {
  int tid = threadIdx.x;
  int w = blockIdx.x * 8 + (tid >> 5);     // 2048 waves
  int lane = tid & 31, g = lane >> 4, r = lane & 15;
  int ct = w & 15, mt = w >> 4;            // mt: 0..127
  int m0 = mt * 32, c0 = ct * 64;

  v8f acc[2][4];
#pragma unroll
  for (int i = 0; i < 2; ++i)
#pragma unroll
    for (int j = 0; j < 4; ++j) acc[i][j] = zero8f();

  for (int kb = 0; kb < CCH; kb += 32) {
    v16h af[2];
#pragma unroll
    for (int i = 0; i < 2; ++i) {
      const _Float16* ap = A + (size_t)(m0 + 16 * i + r) * CCH + kb;
      af[i] = mk16(*(const v8h*)(ap + 16 * g), *(const v8h*)(ap + 16 * g + 8));
    }
#pragma unroll
    for (int j = 0; j < 4; ++j) {
      const _Float16* bp = BT + (size_t)(c0 + 16 * j + r) * CCH + kb + 16 * g;
      v16h bf = *(const v16h*)bp;
      acc[0][j] = wmma_f16(af[0], bf, acc[0][j]);
      acc[1][j] = wmma_f16(af[1], bf, acc[1][j]);
    }
  }

#pragma unroll
  for (int i = 0; i < 2; ++i) {
#pragma unroll
    for (int j = 0; j < 4; ++j) {
#pragma unroll
      for (int v = 0; v < 8; ++v) {
        int row = m0 + 16 * i + v + 8 * g;
        int col = c0 + 16 * j + r;
        float val = acc[i][j][v] + bias[col];
        if (mode == 0) {
          int n = row >> 10, l = row & 1023, h = col >> 5, e = col & 31;
          ((_Float16*)outp)[(((size_t)(n * NHH + h) * LSEQ) + l) * EE + e] =
              (_Float16)val;
        } else if (mode == 1) {
          int n = row >> 10, l = row & 1023, h = col >> 5, e = col & 31;
          ((_Float16*)outp)[((size_t)(n * NHH + h) * EE + e) * LSEQ + l] =
              (_Float16)val;
        } else if (mode == 2) {
          val = fmaxf(val, 0.f);
          ((_Float16*)outp)[(size_t)row * CCH + col] = (_Float16)val;
        } else {
          ((float*)outp)[(size_t)row * CCH + col] = val;
        }
      }
    }
  }
}

// ---------------- positional projection: Qp[nh][l][d] = q[nh][l] . emb[d] ----
__global__ __launch_bounds__(256) void k_posgemm(const _Float16* __restrict__ src,
                                                 const _Float16* __restrict__ embh,
                                                 _Float16* __restrict__ dst) {
  int tid = threadIdx.x;
  int w = blockIdx.x * 8 + (tid >> 5);         // 128 * 64 * 17 waves
  int lane = tid & 31, g = lane >> 4, r = lane & 15;
  int nh = w / (64 * 17);
  int rem = w % (64 * 17);
  int lt = rem / 17, dt = rem % 17;
  int l0 = lt * 16, d0 = dt * 16;

  const _Float16* ap = src + ((size_t)(nh << 10) + l0 + r) * EE;
  v16h af = mk16(*(const v8h*)(ap + 16 * g), *(const v8h*)(ap + 16 * g + 8));
  v16h bf = *(const v16h*)(embh + (size_t)(d0 + r) * EE + 16 * g);
  v8f c = wmma_f16(af, bf, zero8f());

#pragma unroll
  for (int v = 0; v < 8; ++v) {
    int l = l0 + v + 8 * g;
    int d = d0 + r;
    if (d < NDIST)
      dst[((size_t)(nh << 10) + l) * PSTR + d] = (_Float16)c[v];
  }
}

// ---------------- fused disentangled attention (flash-style) -----------------
// one wave owns a 16-row l tile of one (n,h); streams s in 32-wide steps.
// Running max is row-reduced each step (needed before exp); the softmax
// denominator is accumulated per-lane and reduced across lanes once at the end.
__global__ __launch_bounds__(256) void k_attn(const _Float16* __restrict__ qh,
                                              const _Float16* __restrict__ kh,
                                              const _Float16* __restrict__ vT,
                                              const _Float16* __restrict__ Qp,
                                              const _Float16* __restrict__ Kp,
                                              float* __restrict__ t) {
  __shared__ __align__(64) _Float16 Pl[8][16][32];
  int tid = threadIdx.x, ws = tid >> 5, lane = tid & 31;
  int g = lane >> 4, r = lane & 15;
  int w = blockIdx.x * 8 + ws;                 // 8192 waves
  int nh = w >> 6, lt = w & 63;
  int l0 = lt << 4;
  const float temp = 0.17677669529663687f;     // 1/sqrt(32)

  const _Float16* qp_ = qh + ((size_t)(nh << 10) + l0 + r) * EE;
  v16h qf = mk16(*(const v8h*)(qp_ + 16 * g), *(const v8h*)(qp_ + 16 * g + 8));

  v8f O0 = zero8f(), O1 = zero8f();
  float mrow[8], lsum[8];                      // lsum: per-lane partial
#pragma unroll
  for (int v = 0; v < 8; ++v) { mrow[v] = -1e30f; lsum[v] = 0.f; }

  const _Float16* Qpb = Qp + (size_t)(nh << 10) * PSTR;
  const _Float16* Kpb = Kp + (size_t)(nh << 10) * PSTR;

  for (int s0 = 0; s0 < LSEQ; s0 += 32) {
    v16h kb0 = *(const v16h*)(kh + ((size_t)(nh << 10) + s0 + r) * EE + 16 * g);
    v16h kb1 = *(const v16h*)(kh + ((size_t)(nh << 10) + s0 + 16 + r) * EE + 16 * g);
    v8f S0 = wmma_f16(qf, kb0, zero8f());
    v8f S1 = wmma_f16(qf, kb1, zero8f());

    float p0[8], p1[8];
#pragma unroll
    for (int v = 0; v < 8; ++v) {
      int l = l0 + v + 8 * g;
      int sa = s0 + r, sb = s0 + 16 + r;
      int da = l - sa; da = da < -128 ? -128 : (da > 128 ? 128 : da); da += 128;
      int db = l - sb; db = db < -128 ? -128 : (db > 128 ? 128 : db); db += 128;
      float va = (S0[v] + (float)Qpb[(size_t)l * PSTR + da] +
                  (float)Kpb[(size_t)sa * PSTR + da]) * temp;
      float vb = (S1[v] + (float)Qpb[(size_t)l * PSTR + db] +
                  (float)Kpb[(size_t)sb * PSTR + db]) * temp;
      // row max over the 16 lanes of this half-wave (xor<16 stays in half)
      float rm = fmaxf(va, vb);
#pragma unroll
      for (int msk = 1; msk < 16; msk <<= 1) rm = fmaxf(rm, __shfl_xor(rm, msk, 32));
      float mnew = fmaxf(mrow[v], rm);
      float alpha = __expf(mrow[v] - mnew);
      float ea = __expf(va - mnew), eb = __expf(vb - mnew);
      lsum[v] = lsum[v] * alpha + (ea + eb);   // per-lane partial denominator
      mrow[v] = mnew;
      O0[v] *= alpha;
      O1[v] *= alpha;
      p0[v] = ea; p1[v] = eb;
    }

    // C-layout P -> A-layout f16 fragment via LDS round trip
#pragma unroll
    for (int v = 0; v < 8; ++v) {
      Pl[ws][v + 8 * g][r]      = (_Float16)p0[v];
      Pl[ws][v + 8 * g][16 + r] = (_Float16)p1[v];
    }
    asm volatile("s_wait_dscnt 0" ::: "memory");
    v16h pf = mk16(*(const v8h*)(&Pl[ws][r][16 * g]),
                   *(const v8h*)(&Pl[ws][r][16 * g + 8]));

    v16h vb0 = *(const v16h*)(vT + ((size_t)(nh << 5) + r) * LSEQ + s0 + 16 * g);
    v16h vb1 = *(const v16h*)(vT + ((size_t)(nh << 5) + 16 + r) * LSEQ + s0 + 16 * g);
    O0 = wmma_f16(pf, vb0, O0);
    O1 = wmma_f16(pf, vb1, O1);
  }

  int n = nh >> 5, h = nh & 31;
#pragma unroll
  for (int v = 0; v < 8; ++v) {
    // reduce per-lane partial denominators across the 16 row-lanes (once)
    float rs = lsum[v];
#pragma unroll
    for (int msk = 1; msk < 16; msk <<= 1) rs += __shfl_xor(rs, msk, 32);
    int l = l0 + v + 8 * g;
    float inv = 1.0f / rs;
    size_t base = ((size_t)(n << 10) + l) * CCH + (h << 5);
    t[base + r]      = O0[v] * inv;
    t[base + 16 + r] = O1[v] * inv;
  }
}

// ---------------- AddNorm: out = LN(a + res); optional f16 copy + swap -------
__global__ __launch_bounds__(256) void k_addnorm(const float* __restrict__ a,
                                                 const float* __restrict__ res,
                                                 const float* __restrict__ gg,
                                                 const float* __restrict__ bb,
                                                 float* __restrict__ outf,
                                                 _Float16* __restrict__ outh,
                                                 int swap) {
  __shared__ float red[256];
  int row = blockIdx.x, tid = threadIdx.x;
  float vals[4];
  float s = 0.f;
#pragma unroll
  for (int i = 0; i < 4; ++i) {
    int c = tid + i * 256;
    vals[i] = a[(size_t)row * CCH + c] + res[(size_t)row * CCH + c];
    s += vals[i];
  }
  red[tid] = s; __syncthreads();
  for (int off = 128; off > 0; off >>= 1) {
    if (tid < off) red[tid] += red[tid + off];
    __syncthreads();
  }
  float mean = red[0] * (1.0f / 1024.0f);
  __syncthreads();
  float s2 = 0.f;
#pragma unroll
  for (int i = 0; i < 4; ++i) { float d = vals[i] - mean; s2 += d * d; }
  red[tid] = s2; __syncthreads();
  for (int off = 128; off > 0; off >>= 1) {
    if (tid < off) red[tid] += red[tid + off];
    __syncthreads();
  }
  float var = red[0] * (1.0f / 1024.0f);
  float rstd = rsqrtf(var + 1e-5f);
#pragma unroll
  for (int i = 0; i < 4; ++i) {
    int c = tid + i * 256;
    float y = (vals[i] - mean) * rstd * gg[c] + bb[c];
    size_t oidx;
    if (swap) {
      int n = row >> 10, l = row & 1023;
      oidx = ((size_t)l * NB + n) * CCH + c;
    } else {
      oidx = (size_t)row * CCH + c;
    }
    outf[oidx] = y;
    if (outh) outh[(size_t)row * CCH + c] = (_Float16)y;
  }
}

extern "C" void kernel_launch(void* const* d_in, const int* in_sizes, int n_in,
                              void* d_out, int out_size, void* d_ws, size_t ws_size,
                              hipStream_t stream) {
  const float* x     = (const float*)d_in[0];
  const float* Wq    = (const float*)d_in[1];
  const float* bq    = (const float*)d_in[2];
  const float* Wk    = (const float*)d_in[3];
  const float* bk    = (const float*)d_in[4];
  const float* Wv    = (const float*)d_in[5];
  const float* bv    = (const float*)d_in[6];
  const float* demb  = (const float*)d_in[7];
  const float* ln1_g = (const float*)d_in[8];
  const float* ln1_b = (const float*)d_in[9];
  const float* ln2_g = (const float*)d_in[10];
  const float* ln2_b = (const float*)d_in[11];
  const float* W1    = (const float*)d_in[12];
  const float* b1    = (const float*)d_in[13];
  const float* W2    = (const float*)d_in[14];
  const float* b2    = (const float*)d_in[15];

  char* p = (char*)d_ws;
  auto carve = [&](size_t bytes) -> void* {
    void* r = (void*)p;
    p += (bytes + 255) & ~(size_t)255;
    return r;
  };
  _Float16* xh   = (_Float16*)carve((size_t)MROWS * CCH * 2);
  _Float16* WqT  = (_Float16*)carve((size_t)CCH * CCH * 2);
  _Float16* WkT  = (_Float16*)carve((size_t)CCH * CCH * 2);
  _Float16* WvT  = (_Float16*)carve((size_t)CCH * CCH * 2);
  _Float16* W1T  = (_Float16*)carve((size_t)CCH * CCH * 2);
  _Float16* W2T  = (_Float16*)carve((size_t)CCH * CCH * 2);
  _Float16* embh = (_Float16*)carve((size_t)272 * EE * 2);
  _Float16* qh   = (_Float16*)carve((size_t)NB * NHH * LSEQ * EE * 2);
  _Float16* khb  = (_Float16*)carve((size_t)NB * NHH * LSEQ * EE * 2);
  _Float16* vTT  = (_Float16*)carve((size_t)NB * NHH * LSEQ * EE * 2);
  _Float16* Qp   = (_Float16*)carve((size_t)NB * NHH * LSEQ * PSTR * 2);
  _Float16* Kp   = (_Float16*)carve((size_t)NB * NHH * LSEQ * PSTR * 2);
  float*    tbuf = (float*)carve((size_t)MROWS * CCH * 4);
  float*    hbuf = (float*)carve((size_t)MROWS * CCH * 4);
  _Float16* hh   = (_Float16*)carve((size_t)MROWS * CCH * 2);
  _Float16* mid  = (_Float16*)carve((size_t)MROWS * CCH * 2);
  float*    mbuf = (float*)carve((size_t)MROWS * CCH * 4);

  // stage f16 operands
  k_cvt<<<(MROWS * CCH) / 256, 256, 0, stream>>>(x, xh, MROWS * CCH, MROWS * CCH);
  k_cvt<<<34, 256, 0, stream>>>(demb, embh, NDIST * EE, 272 * EE);
  k_trans<<<1024, 256, 0, stream>>>(Wq, WqT);
  k_trans<<<1024, 256, 0, stream>>>(Wk, WkT);
  k_trans<<<1024, 256, 0, stream>>>(Wv, WvT);
  k_trans<<<1024, 256, 0, stream>>>(W1, W1T);
  k_trans<<<1024, 256, 0, stream>>>(W2, W2T);

  // QKV projections (WMMA, 32x64 tiles per wave)
  k_gemm<<<256, 256, 0, stream>>>(xh, WqT, bq, qh, 0);
  k_gemm<<<256, 256, 0, stream>>>(xh, WkT, bk, khb, 0);
  k_gemm<<<256, 256, 0, stream>>>(xh, WvT, bv, vTT, 1);

  // positional projections (WMMA)
  k_posgemm<<<17408, 256, 0, stream>>>(qh, embh, Qp);
  k_posgemm<<<17408, 256, 0, stream>>>(khb, embh, Kp);

  // fused attention (WMMA + online softmax + gathered positional terms)
  k_attn<<<1024, 256, 0, stream>>>(qh, khb, vTT, Qp, Kp, tbuf);

  // AddNorm 1
  k_addnorm<<<MROWS, 256, 0, stream>>>(tbuf, x, ln1_g, ln1_b, hbuf, hh, 0);

  // MLP (WMMA)
  k_gemm<<<256, 256, 0, stream>>>(hh, W1T, b1, mid, 2);
  k_gemm<<<256, 256, 0, stream>>>(mid, W2T, b2, mbuf, 3);

  // AddNorm 2 + (n,l,c) -> (l,n,c) swap into d_out
  k_addnorm<<<MROWS, 256, 0, stream>>>(mbuf, hbuf, ln2_g, ln2_b, (float*)d_out,
                                       nullptr, 1);
}
